// BasePllay_05_38276748542092
// MI455X (gfx1250) — matrix operand order repaired and checked by reading, not compiled, not run on hardware
//
#include <hip/hip_runtime.h>
#include <hip/hip_bf16.h>
#include <math.h>

// ---------------------------------------------------------------------------
// Pipeline: dtm (histogram-based, no sort) -> ph0 (bitonic sort + serial UF)
//           -> landscape (top-2 reduction) -> head (WMMA f16 GEMMs, LDS-staged)
// ---------------------------------------------------------------------------

#define Hh 64
#define Ww 64
#define Nn (Hh * Ww)            // 4096
#define R2MAX 7939              // 63^2 + 63^2 + 1 bins
#define Eh 4032                 // horizontal edges 64*63
#define Ee 8064                 // total edges
#define EP1 8065                // edges + essential bar
#define SORTN 8192              // padded pow2 for bitonic
#define Tt 25

typedef __attribute__((ext_vector_type(16))) _Float16 v16h;
typedef __attribute__((ext_vector_type(8)))  float    v8f;

// ---------------------------------------------------------------------------
// Kernel 1: weighted distance-to-measure. One block per (point, batch).
// bins[r2] accumulates weight mass at squared lattice distance r2; the
// sorted-cumsum of the reference telescopes to min(thr,c_k)-min(thr,c_{k-1}),
// which is order-invariant within equal-distance groups, so the histogram is
// exact.
// ---------------------------------------------------------------------------
__global__ __launch_bounds__(256)
void dtm_kernel(const float* __restrict__ input, float* __restrict__ f_out) {
    const int bat = blockIdx.y;
    const int i   = blockIdx.x;
    const int tid = threadIdx.x;

    __shared__ float bins[R2MAX];
    __shared__ float red[256];

    for (int k = tid; k < R2MAX; k += 256) bins[k] = 0.0f;
    __syncthreads();

    const float* w = input + (size_t)bat * Nn;
    const int ri = i >> 6, ci = i & 63;

    float wloc = 0.0f;
    for (int j = tid; j < Nn; j += 256) {
        float wv = w[j];
        int dr = ri - (j >> 6);
        int dc = ci - (j & 63);
        atomicAdd(&bins[dr * dr + dc * dc], wv);   // ds_add_f32
        wloc += wv;
    }
    red[tid] = wloc;
    __syncthreads();
    for (int off = 128; off > 0; off >>= 1) {
        if (tid < off) red[tid] += red[tid + off];
        __syncthreads();
    }
    const float thr = 0.05f * red[0];
    __syncthreads();

    // Per-thread chunk of 32 bins -> partial sums -> exclusive scan -> walk.
    const int k0 = tid * 32;
    const int k1 = (k0 + 32 < R2MAX) ? (k0 + 32) : R2MAX;
    float psum = 0.0f;
    for (int k = k0; k < k1; ++k) psum += bins[k];

    red[tid] = psum;
    __syncthreads();
    for (int off = 1; off < 256; off <<= 1) {
        float v = (tid >= off) ? red[tid - off] : 0.0f;
        __syncthreads();
        red[tid] += v;
        __syncthreads();
    }
    float c  = red[tid] - psum;      // exclusive prefix (mass before my bins)
    __syncthreads();

    float mp = fminf(thr, c);
    float S  = 0.0f;
    for (int k = k0; k < k1; ++k) {
        float cn  = c + bins[k];
        float mnn = fminf(thr, cn);
        S += (float)k * (mnn - mp);
        mp = mnn;
        c  = cn;
    }
    red[tid] = S;
    __syncthreads();
    for (int off = 128; off > 0; off >>= 1) {
        if (tid < off) red[tid] += red[tid + off];
        __syncthreads();
    }
    if (tid == 0) {
        const float s2 = (224.0f / 63.0f) * (224.0f / 63.0f);
        f_out[(size_t)bat * Nn + i] = sqrtf(red[0] * s2 / thr);
    }
}

// ---------------------------------------------------------------------------
// Kernel 2: 0-dim persistence. One block per batch.
// Keys (wgt_bits<<32 | edge_idx) reproduce JAX stable argsort (f >= 0, so
// float bit patterns are order-preserving). Bitonic sort on 8192 keys in
// workspace global memory, then one thread runs the elder-rule union-find
// (path compression is output-equivalent to the reference's full relabel).
// ---------------------------------------------------------------------------
__global__ __launch_bounds__(256)
void ph0_kernel(const float* __restrict__ f_all,
                unsigned long long* __restrict__ keys_all,
                float* __restrict__ bb, float* __restrict__ bd,
                int* __restrict__ bv) {
    const int bat = blockIdx.x;
    const int tid = threadIdx.x;

    __shared__ float fsh[Nn];
    __shared__ int   label[Nn];
    __shared__ float redmin[256], redmax[256];

    const float* f = f_all + (size_t)bat * Nn;
    unsigned long long* keys = keys_all + (size_t)bat * SORTN;

    for (int i = tid; i < Nn; i += 256) { fsh[i] = f[i]; label[i] = i; }
    __syncthreads();

    float mn = 3.0e38f, mx = -3.0e38f;
    for (int i = tid; i < Nn; i += 256) {
        mn = fminf(mn, fsh[i]);
        mx = fmaxf(mx, fsh[i]);
    }
    redmin[tid] = mn; redmax[tid] = mx;
    __syncthreads();
    for (int off = 128; off > 0; off >>= 1) {
        if (tid < off) {
            redmin[tid] = fminf(redmin[tid], redmin[tid + off]);
            redmax[tid] = fmaxf(redmax[tid], redmax[tid + off]);
        }
        __syncthreads();
    }

    // Build composite sort keys.
    for (int e = tid; e < SORTN; e += 256) {
        unsigned long long key;
        if (e < Ee) {
            int eu, ev;
            if (e < Eh) { int r = e / 63, cc = e % 63; eu = r * 64 + cc; ev = eu + 1; }
            else        { eu = e - Eh; ev = eu + 64; }
            float wgt = fmaxf(fsh[eu], fsh[ev]);
            unsigned int ub = __float_as_uint(wgt);
            key = ((unsigned long long)ub << 32) | (unsigned int)e;
        } else {
            key = ~0ULL;
        }
        keys[e] = key;
    }
    __syncthreads();

    // Intra-block bitonic sort over global-memory keys.
    for (int k = 2; k <= SORTN; k <<= 1) {
        for (int j = k >> 1; j > 0; j >>= 1) {
            for (int i = tid; i < SORTN; i += 256) {
                int ixj = i ^ j;
                if (ixj > i) {
                    unsigned long long a = keys[i];
                    unsigned long long b = keys[ixj];
                    bool up = ((i & k) == 0);
                    if ((up && a > b) || (!up && a < b)) {
                        keys[i] = b;
                        keys[ixj] = a;
                    }
                }
            }
            __syncthreads();
        }
    }

    // Sequential elder-rule scan (thread 0).
    if (tid == 0) {
        float* Bb = bb + (size_t)bat * EP1;
        float* Bd = bd + (size_t)bat * EP1;
        int*   Bv = bv + (size_t)bat * EP1;
        for (int t = 0; t < Ee; ++t) {
            int e = (int)(unsigned int)(keys[t] & 0xffffffffULL);
            int eu, ev;
            if (e < Eh) { int r = e / 63, cc = e % 63; eu = r * 64 + cc; ev = eu + 1; }
            else        { eu = e - Eh; ev = eu + 64; }
            int ru = eu;
            while (label[ru] != ru) { label[ru] = label[label[ru]]; ru = label[ru]; }
            int rv = ev;
            while (label[rv] != rv) { label[rv] = label[label[rv]]; rv = label[rv]; }
            float bu = fsh[ru], bvv = fsh[rv];
            bool  ue = (bu < bvv) || ((bu == bvv) && (ru <= rv));
            int elder   = ue ? ru : rv;
            int younger = ue ? rv : ru;
            bool merge  = (ru != rv);
            Bb[t] = fsh[younger];
            Bd[t] = fmaxf(fsh[eu], fsh[ev]);
            Bv[t] = merge ? 1 : 0;
            if (merge) label[younger] = elder;
        }
        Bb[Ee] = redmin[0];
        Bd[Ee] = redmax[0];
        Bv[Ee] = 1;
    }
}

// ---------------------------------------------------------------------------
// Kernel 3: landscape top-2 per (batch, t). One block per (t, batch).
// ---------------------------------------------------------------------------
__global__ __launch_bounds__(256)
void landscape_kernel(const float* __restrict__ bb, const float* __restrict__ bd,
                      const int* __restrict__ bv, float* __restrict__ land) {
    const int t   = blockIdx.x;
    const int bat = blockIdx.y;
    const int tid = threadIdx.x;
    const float ts = (45.0f / 24.0f) * (float)t;

    const float* Bb = bb + (size_t)bat * EP1;
    const float* Bd = bd + (size_t)bat * EP1;
    const int*   Bv = bv + (size_t)bat * EP1;

    float m1 = 0.0f, m2 = 0.0f;   // all candidates are >= 0 after masking
    for (int i = tid; i < EP1; i += 256) {
        float tri = fminf(ts - Bb[i], Bd[i] - ts);
        float v   = Bv[i] ? fmaxf(tri, 0.0f) : 0.0f;
        if (v > m1) { m2 = m1; m1 = v; }
        else if (v > m2) { m2 = v; }
    }
    __shared__ float s1[256], s2[256];
    s1[tid] = m1; s2[tid] = m2;
    __syncthreads();
    for (int off = 128; off > 0; off >>= 1) {
        if (tid < off) {
            float a1 = s1[tid], a2 = s2[tid];
            float c1 = s1[tid + off], c2 = s2[tid + off];
            s1[tid] = fmaxf(a1, c1);
            s2[tid] = fmaxf(fminf(a1, c1), fmaxf(a2, c2));
        }
        __syncthreads();
    }
    if (tid == 0) {
        land[(size_t)bat * 50 + t]      = s1[0];   // k = 0
        land[(size_t)bat * 50 + 25 + t] = s2[0];   // k = 1
    }
}

// ---------------------------------------------------------------------------
// Kernel 4: head. One wave. All operands staged once into zero-padded f16
// LDS tiles, then fed to v_wmma_f32_16x16x32_f16 with unconditional ds reads
// (no divergent guarded global loads). Fragment element maps per ISA 7.12.2:
//   A (16x32 f16): M = lane&15,  K = (e>>3)*16 + (e&7) + 8*(lane>>4)
//                  -> two contiguous 8-half runs per lane
//   B (32x16 f16): N = lane&15,  K = (e>>3)*16 + (e&7) + 8*(lane>>4)
//                  -> two contiguous 8-half runs per lane (row-major weights)
//   C/D (16x16 f32): N = lane&15, M = r + 8*(lane>>4)
// x = land @ Wg^T + bg ; signal = sum_b |x| ; out = relu(x) @ Wfc^T + bfc.
// ---------------------------------------------------------------------------
__global__ __launch_bounds__(32)
void head_kernel(const float* __restrict__ land, const float* __restrict__ Wg,
                 const float* __restrict__ bg, const float* __restrict__ Wfc,
                 const float* __restrict__ bfc, float* __restrict__ out,
                 int Bsz) {
    const int lane = threadIdx.x;
    const int Ml   = lane & 15;
    const int grp  = lane >> 4;

    // Zero-padded f16 operand tiles ([row][K], K padded to 64).
    __shared__ _Float16 Apad[16][64];   // land rows (re-used for relu(x))
    __shared__ _Float16 Bg[64][64];     // Wg[n][k], n padded to 64
    __shared__ _Float16 Bf[16][64];     // Wfc[n][k], n padded to 16
    __shared__ float    xsh[2 * 50];

    for (int idx = lane; idx < 16 * 64; idx += 32) {
        int m = idx >> 6, k = idx & 63;
        Apad[m][k] = (_Float16)((m < Bsz && k < 50) ? land[m * 50 + k] : 0.0f);
    }
    for (int idx = lane; idx < 64 * 64; idx += 32) {
        int n = idx >> 6, k = idx & 63;
        Bg[n][k] = (_Float16)((n < 50 && k < 50) ? Wg[n * 50 + k] : 0.0f);
    }
    for (int idx = lane; idx < 16 * 64; idx += 32) {
        int n = idx >> 6, k = idx & 63;
        Bf[n][k] = (_Float16)((n < 10 && k < 50) ? Wfc[n * 50 + k] : 0.0f);
    }
    __syncthreads();

    // ---- GEMM1: (B x 50) @ (50 x 50)^T, K padded to 64, N tiled by 16 ----
#pragma unroll
    for (int ntile = 0; ntile < 4; ++ntile) {
        v8f acc = {};
#pragma unroll
        for (int kc = 0; kc < 2; ++kc) {
            v16h a, b;
#pragma unroll
            for (int e = 0; e < 16; ++e) {
                int Ka = kc * 32 + ((e >> 3) << 4) + (e & 7) + (grp << 3);
                a[e] = Apad[Ml][Ka];
                int Kb = kc * 32 + ((e >> 3) << 4) + (e & 7) + (grp << 3);
                b[e] = Bg[ntile * 16 + Ml][Kb];
            }
            acc = __builtin_amdgcn_wmma_f32_16x16x32_f16(
                false, a, false, b, (short)0, acc, false, false);
        }
#pragma unroll
        for (int r = 0; r < 8; ++r) {
            int M  = r + (grp << 3);
            int ng = ntile * 16 + Ml;
            if (M < Bsz && ng < 50) xsh[M * 50 + ng] = acc[r] + bg[ng];
        }
    }
    __syncthreads();

    // ---- signal = sum over batch of |x| ----
    for (int j = lane; j < 50; j += 32) {
        float s = 0.0f;
        for (int m = 0; m < Bsz; ++m) s += fabsf(xsh[m * 50 + j]);
        out[Bsz * 10 + j] = s;
    }
    __syncthreads();

    // Re-stage relu(x) into Apad.
    for (int idx = lane; idx < 16 * 64; idx += 32) {
        int m = idx >> 6, k = idx & 63;
        Apad[m][k] = (_Float16)((m < Bsz && k < 50) ? fmaxf(xsh[m * 50 + k], 0.0f)
                                                    : 0.0f);
    }
    __syncthreads();

    // ---- GEMM2: relu(x) (B x 50) @ (10 x 50)^T ----
    v8f acc = {};
#pragma unroll
    for (int kc = 0; kc < 2; ++kc) {
        v16h a, b;
#pragma unroll
        for (int e = 0; e < 16; ++e) {
            int Ka = kc * 32 + ((e >> 3) << 4) + (e & 7) + (grp << 3);
            a[e] = Apad[Ml][Ka];
            int Kb = kc * 32 + ((e >> 3) << 4) + (e & 7) + (grp << 3);
            b[e] = Bf[Ml][Kb];
        }
        acc = __builtin_amdgcn_wmma_f32_16x16x32_f16(
            false, a, false, b, (short)0, acc, false, false);
    }
#pragma unroll
    for (int r = 0; r < 8; ++r) {
        int M = r + (grp << 3);
        if (M < Bsz && Ml < 10) out[M * 10 + Ml] = acc[r] + bfc[Ml];
    }
}

// ---------------------------------------------------------------------------
// Launch. Inputs: input[B*4096], Wg[50*50], bg[50], Wfc[10*50], bfc[10].
// Output: [B*10 output | 50 signal]. Workspace ~350 KB.
// ---------------------------------------------------------------------------
extern "C" void kernel_launch(void* const* d_in, const int* in_sizes, int n_in,
                              void* d_out, int out_size, void* d_ws, size_t ws_size,
                              hipStream_t stream) {
    (void)n_in; (void)out_size; (void)ws_size;
    const float* input = (const float*)d_in[0];
    const float* Wg    = (const float*)d_in[1];
    const float* bg    = (const float*)d_in[2];
    const float* Wfc   = (const float*)d_in[3];
    const float* bfc   = (const float*)d_in[4];
    float* out = (float*)d_out;

    const int B = in_sizes[0] / Nn;   // 2 in the reference setup

    char* ws = (char*)d_ws;
    size_t off = 0;
    float* f_ws = (float*)(ws + off);                 off += (size_t)B * Nn * sizeof(float);
    off = (off + 15) & ~(size_t)15;
    unsigned long long* keys = (unsigned long long*)(ws + off);
                                                      off += (size_t)B * SORTN * sizeof(unsigned long long);
    float* bb = (float*)(ws + off);                   off += (size_t)B * EP1 * sizeof(float);
    float* bd = (float*)(ws + off);                   off += (size_t)B * EP1 * sizeof(float);
    int*   bv = (int*)(ws + off);                     off += (size_t)B * EP1 * sizeof(int);
    off = (off + 15) & ~(size_t)15;
    float* land = (float*)(ws + off);

    dtm_kernel<<<dim3(Nn, B), 256, 0, stream>>>(input, f_ws);
    ph0_kernel<<<dim3(B), 256, 0, stream>>>(f_ws, keys, bb, bd, bv);
    landscape_kernel<<<dim3(Tt, B), 256, 0, stream>>>(bb, bd, bv, land);
    head_kernel<<<dim3(1), 32, 0, stream>>>(land, Wg, bg, Wfc, bfc, out, B);
}
